// Net7_81595788689560
// MI455X (gfx1250) — compile-verified, matrix operand order checked by reading, not run on hardware
//
#include <hip/hip_runtime.h>
#include <hip/hip_bf16.h>

// ---------------------------------------------------------------------------
// ChebConv GNN pipeline for MI455X (gfx1250, wave32).
// Dense GEMMs run on v_wmma_f32_16x16x32_bf16 (f32 accumulate). The A tile is
// DMA'd global->LDS with gfx1250 async loads (ASYNCcnt, LDS double-buffer) so
// the copy overlaps the WMMA pipe without consuming VGPRs; B tile uses a
// register-staged bf16 transpose. Sparse Laplacian matvecs use hardware fp32
// global atomics; working set (~85MB) is L2-resident (192MB L2).
// ---------------------------------------------------------------------------

typedef __bf16 bf16_t;
typedef __attribute__((ext_vector_type(16))) __bf16 v16bf;
typedef __attribute__((ext_vector_type(8)))  __bf16 v8bf;
typedef __attribute__((ext_vector_type(4)))  __bf16 v4bf;
typedef __attribute__((ext_vector_type(8)))  float  v8f;

namespace {
constexpr int   Nn  = 20000;
constexpr int   Ee  = 320000;
constexpr int   FIN = 128;
constexpr int   D1c = 256;
constexpr int   D2c = 128;
constexpr int   D3c = 64;
constexpr int   Kc  = 10;
}

#define TM 128
#define TN 64
#define TK 32

// ------------------------------- elementwise -------------------------------

__global__ void fill_kernel(float* __restrict__ p, float v, long long n) {
  long long i = (long long)blockIdx.x * blockDim.x + threadIdx.x;
  if (i < n) p[i] = v;
}

__global__ void neg_copy_kernel(const float* __restrict__ in, float* __restrict__ out,
                                long long n) {
  long long i = (long long)blockIdx.x * blockDim.x + threadIdx.x;
  if (i < n) out[i] = -in[i];
}

__global__ void relu_kernel(float* __restrict__ p, long long n) {
  long long i = (long long)blockIdx.x * blockDim.x + threadIdx.x;
  if (i < n) p[i] = fmaxf(p[i], 0.0f);
}

__global__ void bias_init_kernel(float* __restrict__ out, const float* __restrict__ b,
                                 int mask, long long n) {
  long long i = (long long)blockIdx.x * blockDim.x + threadIdx.x;
  if (i < n) out[i] = b[(int)(i & mask)];
}

// ------------------------------ graph precompute ---------------------------

__global__ void deg_kernel(const long long* __restrict__ dst, float* __restrict__ deg,
                           int E) {
  int e = blockIdx.x * blockDim.x + threadIdx.x;
  if (e < E)
    __hip_atomic_fetch_add(&deg[(int)dst[e]], 1.0f, __ATOMIC_RELAXED,
                           __HIP_MEMORY_SCOPE_AGENT);
}

__global__ void dinv_kernel(const float* __restrict__ deg, float* __restrict__ dinv,
                            int n) {
  int i = blockIdx.x * blockDim.x + threadIdx.x;
  if (i < n) {
    float d = deg[i];
    dinv[i] = d > 0.0f ? rsqrtf(fmaxf(d, 1.0f)) : 0.0f;
  }
}

__global__ void norm_kernel(const long long* __restrict__ src,
                            const long long* __restrict__ dst,
                            const float* __restrict__ dinv, float* __restrict__ nw,
                            int E) {
  int e = blockIdx.x * blockDim.x + threadIdx.x;
  if (e < E) nw[e] = -dinv[(int)src[e]] * dinv[(int)dst[e]];
}

// ------------------------------ sparse matvec ------------------------------
// out[dst] += alpha * norm[e] * in[src]   (out pre-initialized by caller)

__global__ void spmv_kernel(const float* __restrict__ in, float* __restrict__ out,
                            const long long* __restrict__ src,
                            const long long* __restrict__ dst,
                            const float* __restrict__ nw, float alpha, int l2d4,
                            long long total) {
  long long idx = (long long)blockIdx.x * blockDim.x + threadIdx.x;
  if (idx >= total) return;
  int e = (int)(idx >> l2d4);
  int q = (int)(idx & ((1 << l2d4) - 1));
  float w = alpha * nw[e];
  long long D = 4LL << l2d4;
  const float4 v = *(const float4*)(in + (long long)src[e] * D + (long long)q * 4);
  float* o = out + (long long)dst[e] * D + (long long)q * 4;
  __hip_atomic_fetch_add(o + 0, w * v.x, __ATOMIC_RELAXED, __HIP_MEMORY_SCOPE_AGENT);
  __hip_atomic_fetch_add(o + 1, w * v.y, __ATOMIC_RELAXED, __HIP_MEMORY_SCOPE_AGENT);
  __hip_atomic_fetch_add(o + 2, w * v.z, __ATOMIC_RELAXED, __HIP_MEMORY_SCOPE_AGENT);
  __hip_atomic_fetch_add(o + 3, w * v.w, __ATOMIC_RELAXED, __HIP_MEMORY_SCOPE_AGENT);
}

// ------------------------------ WMMA GEMM ----------------------------------
// C(MxN) += A(MxK) @ B(KxN), fp32 in/out, bf16 WMMA 16x16x32 compute.
// Block = 8 waves, 128x64 macro-tile; wave w owns rows w*16..w*16+15, four
// 16x16 accumulators across the 64-wide N strip.
// A tile: async global->LDS DMA (f32, LDS double-buffered, ASYNCcnt), bf16
// conversion during fragment build. B tile: register-pipelined bf16 transpose.

__global__ __launch_bounds__(256) void gemm_bf16_wmma(
    const float* __restrict__ A, const float* __restrict__ B, float* __restrict__ C,
    int M, int Kd, int Nc) {
  __shared__ __align__(16) float  ldsAf[2][TM][TK];  // 2 x 16 KB (f32, async dst)
  __shared__ __align__(16) bf16_t ldsB[TN][TK];      // 4 KB, B transposed: [n][k]

  const int tid  = threadIdx.x;
  const int wave = tid >> 5;
  const int lane = tid & 31;
  const int m0 = blockIdx.x * TM;
  const int n0 = blockIdx.y * TN;
  const int rt   = lane & 15;  // row (A) / col (B) within 16-tile
  const int half = lane >> 4;  // ISA lane-half selects K sub-pattern

  // Per-thread staging coordinates (constant across K-steps):
  //  A: 4 async b128 chunks; chunk i covers row ra0+32*i, cols c4a..c4a+3.
  //  B: 2 chunks of 4 column-wise floats; chunk i at col nb[i], rows kb[i]..+3
  //     (lane-consecutive nb -> coalesced loads; contiguous [n][k] b64 store).
  const int c4a = (tid & 7) * 4;
  const int ra0 = tid >> 3;
  int nb[2], kb[2];
#pragma unroll
  for (int i = 0; i < 2; ++i) {
    int idx = tid + 256 * i;
    nb[i] = idx & 63;
    kb[i] = (idx >> 6) * 4;
  }

  v8f acc[4] = {v8f{}, v8f{}, v8f{}, v8f{}};

  float vb[2][4];

  // Issue this thread's share of the A-tile DMA (4 x async b128 -> LDS).
  auto issue_async_A = [&](int kk, int buf) {
#pragma unroll
    for (int i = 0; i < 4; ++i) {
      int gr = m0 + ra0 + 32 * i;
      if (gr >= M) gr = M - 1;  // clamp; stores are masked
      unsigned voff = ((unsigned)gr * (unsigned)Kd + (unsigned)(kk + c4a)) * 4u;
      // Low 32 bits of a generic LDS pointer == LDS byte offset (aperture in
      // the high half), which is what the async VDST operand wants.
      unsigned ldst = (unsigned)(unsigned long long)&ldsAf[buf][ra0 + 32 * i][c4a];
      asm volatile("global_load_async_to_lds_b128 %0, %1, %2"
                   :: "v"(ldst), "v"(voff), "s"(A)
                   : "memory");
    }
  };

  // Gather one K-tile of B into registers (coalesced across the wave).
  auto load_tileB = [&](int kk) {
#pragma unroll
    for (int i = 0; i < 2; ++i) {
#pragma unroll
      for (int j = 0; j < 4; ++j)
        vb[i][j] = B[(size_t)(kk + kb[i] + j) * Nc + (n0 + nb[i])];
    }
  };

  issue_async_A(0, 0);
  load_tileB(0);

  int ibuf = 0;
  for (int kk = 0; kk < Kd; kk += TK, ibuf ^= 1) {
    // Store current B registers -> bf16 LDS (b64 stores).
#pragma unroll
    for (int i = 0; i < 2; ++i) {
      v4bf p = {(bf16_t)vb[i][0], (bf16_t)vb[i][1], (bf16_t)vb[i][2],
                (bf16_t)vb[i][3]};
      *(v4bf*)&ldsB[nb[i]][kb[i]] = p;
    }

    // Kick off next A tile DMA into the other LDS buffer, then retire the
    // current tile's 4 async loads (they complete in order).
    if (kk + TK < Kd) {
      issue_async_A(kk + TK, ibuf ^ 1);
      asm volatile("s_wait_asynccnt 0x4" ::: "memory");
    } else {
      asm volatile("s_wait_asynccnt 0x0" ::: "memory");
    }
    __syncthreads();  // all waves' DMA + B stores now visible

    // Prefetch next B tile into registers; overlaps with ds reads + WMMA.
    if (kk + TK < Kd) load_tileB(kk + TK);

    // A fragment: lane holds row (wave*16 + rt); K pattern per ISA:
    // lanes 0-15 -> K {0..7,16..23}, lanes 16-31 -> K {8..15,24..31}
    // => four contiguous 16B f32 chunks per lane, converted to bf16 here.
    const float4* paf = (const float4*)&ldsAf[ibuf][wave * 16 + rt][half * 8];
    float4 a0 = paf[0], a1 = paf[1];   // K = half*8 .. half*8+7
    float4 a2 = paf[4], a3 = paf[5];   // K = 16+half*8 .. 16+half*8+7
    v16bf afrag = {
        (bf16_t)a0.x, (bf16_t)a0.y, (bf16_t)a0.z, (bf16_t)a0.w,
        (bf16_t)a1.x, (bf16_t)a1.y, (bf16_t)a1.z, (bf16_t)a1.w,
        (bf16_t)a2.x, (bf16_t)a2.y, (bf16_t)a2.z, (bf16_t)a2.w,
        (bf16_t)a3.x, (bf16_t)a3.y, (bf16_t)a3.z, (bf16_t)a3.w};

#pragma unroll
    for (int t = 0; t < 4; ++t) {
      const v8bf* pb = (const v8bf*)&ldsB[t * 16 + rt][half * 8];
      v8bf blo = pb[0];
      v8bf bhi = pb[2];
      v16bf bfrag;
#pragma unroll
      for (int i = 0; i < 8; ++i) { bfrag[i] = blo[i]; bfrag[i + 8] = bhi[i]; }
      acc[t] = __builtin_amdgcn_wmma_f32_16x16x32_bf16(
          false, afrag, false, bfrag, (short)0, acc[t], false, false);
    }
    __syncthreads();
  }

  // C layout: VGPR i -> row i + half*8, col = lane&15 (per ISA 16x16 f32 C/D).
#pragma unroll
  for (int t = 0; t < 4; ++t) {
    int n = n0 + t * 16 + rt;
#pragma unroll
    for (int i = 0; i < 8; ++i) {
      int m = m0 + wave * 16 + i + half * 8;
      if (m < M) {
        size_t off = (size_t)m * Nc + n;
        C[off] += acc[t][i];
      }
    }
  }
}

// ------------------------------ batch norm ---------------------------------

__global__ void bn_stats_kernel(const float* __restrict__ h, float* __restrict__ s,
                                float* __restrict__ s2, int M, int D, int rpb,
                                int pre_relu) {
  int c = threadIdx.x;           // blockDim == D (coalesced across columns)
  int r0 = blockIdx.x * rpb;
  int r1 = min(r0 + rpb, M);
  float a = 0.0f, a2 = 0.0f;
  for (int r = r0; r < r1; ++r) {
    float v = h[(long long)r * D + c];
    if (pre_relu) v = fmaxf(v, 0.0f);
    a += v;
    a2 += v * v;
  }
  __hip_atomic_fetch_add(&s[c], a, __ATOMIC_RELAXED, __HIP_MEMORY_SCOPE_AGENT);
  __hip_atomic_fetch_add(&s2[c], a2, __ATOMIC_RELAXED, __HIP_MEMORY_SCOPE_AGENT);
}

__global__ void bn_finalize_kernel(const float* __restrict__ s,
                                   const float* __restrict__ s2,
                                   float* __restrict__ mean, float* __restrict__ inv,
                                   int M) {
  int c = threadIdx.x;
  float m = s[c] / (float)M;
  float v = s2[c] / (float)M - m * m;   // biased variance (matches reference)
  mean[c] = m;
  inv[c] = rsqrtf(v + 1e-5f);
}

__global__ void bn_apply_kernel(float* __restrict__ h, const float* __restrict__ mean,
                                const float* __restrict__ inv,
                                const float* __restrict__ g,
                                const float* __restrict__ b, int mask, long long n,
                                int pre, int post) {
  long long i = (long long)blockIdx.x * blockDim.x + threadIdx.x;
  if (i >= n) return;
  int c = (int)(i & mask);
  float v = h[i];
  if (pre) v = fmaxf(v, 0.0f);
  v = (v - mean[c]) * inv[c] * g[c] + b[c];
  if (post) v = fmaxf(v, 0.0f);
  h[i] = v;
}

// ------------------------------ final head ---------------------------------
// out[r] = dot(h[r, 0:64], w) + b   (one wave per row, shuffle reduction)

__global__ void fc3_kernel(const float* __restrict__ h, const float* __restrict__ w,
                           const float* __restrict__ b, float* __restrict__ out,
                           int M) {
  int row = blockIdx.x * 8 + (threadIdx.x >> 5);
  int lane = threadIdx.x & 31;
  if (row >= M) return;
  float v = h[(long long)row * 64 + lane] * w[lane]
          + h[(long long)row * 64 + 32 + lane] * w[32 + lane];
  for (int off = 16; off > 0; off >>= 1) v += __shfl_down(v, off, 32);
  if (lane == 0) out[row] = v + b[0];
}

// ------------------------------ host driver --------------------------------

static inline unsigned gcount(long long n, int b) {
  return (unsigned)((n + b - 1) / b);
}

extern "C" void kernel_launch(void* const* d_in, const int* in_sizes, int n_in,
                              void* d_out, int out_size, void* d_ws, size_t ws_size,
                              hipStream_t stream) {
  const float*     x       = (const float*)d_in[0];
  const long long* ei      = (const long long*)d_in[1];
  const float*     conv1_w = (const float*)d_in[2];
  const float*     convs_w = (const float*)d_in[3];
  const float*     convs_b = (const float*)d_in[4];
  const float*     bn1_g   = (const float*)d_in[5];
  const float*     bn1_b   = (const float*)d_in[6];
  const float*     fc1_w   = (const float*)d_in[7];
  const float*     fc1_b   = (const float*)d_in[8];
  const float*     bn2_g   = (const float*)d_in[9];
  const float*     bn2_b   = (const float*)d_in[10];
  const float*     fc2_w   = (const float*)d_in[11];
  const float*     fc2_b   = (const float*)d_in[12];
  const float*     bn3_g   = (const float*)d_in[13];
  const float*     bn3_b   = (const float*)d_in[14];
  const float*     fc3_w   = (const float*)d_in[15];
  const float*     fc3_b   = (const float*)d_in[16];

  const long long* srcp = ei;
  const long long* dstp = ei + Ee;

  // Workspace carve-up (~84 MB, fits L2-resident pipeline).
  float* wsf   = (float*)d_ws;
  float* deg   = wsf;  wsf += Nn;
  float* dinv  = wsf;  wsf += Nn;
  float* nw    = wsf;  wsf += Ee;
  float* ssum  = wsf;  wsf += 256;
  float* ssq   = wsf;  wsf += 256;
  float* smean = wsf;  wsf += 256;
  float* sinvv = wsf;  wsf += 256;
  float* B1 = wsf;  wsf += (long long)Nn * 256;
  float* B2 = wsf;  wsf += (long long)Nn * 256;
  float* H0 = wsf;  wsf += (long long)Nn * 256;
  float* H1 = wsf;

  auto fill = [&](float* p, float v, long long n) {
    fill_kernel<<<gcount(n, 256), 256, 0, stream>>>(p, v, n);
  };
  auto gemm = [&](const float* A, const float* B, float* C, int M, int Kd, int Nc) {
    dim3 g((unsigned)((M + TM - 1) / TM), (unsigned)(Nc / TN));
    gemm_bf16_wmma<<<g, 256, 0, stream>>>(A, B, C, M, Kd, Nc);
  };
  auto spmv = [&](const float* in, float* out, float alpha, int D) {
    int l2 = (D == 256) ? 6 : (D == 128 ? 5 : 4);
    long long total = (long long)Ee << l2;
    spmv_kernel<<<gcount(total, 256), 256, 0, stream>>>(in, out, srcp, dstp, nw,
                                                        alpha, l2, total);
  };
  auto bn = [&](float* h, int D, const float* g, const float* b, int pre, int post) {
    fill(ssum, 0.0f, 256);
    fill(ssq, 0.0f, 256);
    const int rpb = 256;
    bn_stats_kernel<<<gcount(Nn, rpb), D, 0, stream>>>(h, ssum, ssq, Nn, D, rpb, pre);
    bn_finalize_kernel<<<1, D, 0, stream>>>(ssum, ssq, smean, sinvv, Nn);
    bn_apply_kernel<<<gcount((long long)Nn * D, 256), 256, 0, stream>>>(
        h, smean, sinvv, g, b, D - 1, (long long)Nn * D, pre, post);
  };
  // Chebyshev layer: accum (pre-initialized) += sum_k T_k @ W_k.
  // T buffers rotate through {r0, r1, r2}; r2 may alias `in` (dead after k=2).
  auto cheb = [&](const float* in, int Din, int Dout, const float* Wb,
                  long long wstride, float* accum, float* r0, float* r1, float* r2) {
    long long nT = (long long)Nn * Din;
    gemm(in, Wb, accum, Nn, Din, Dout);                     // T0 = in
    fill(r0, 0.0f, nT);
    spmv(in, r0, 1.0f, Din);                                // T1 = L^ T0
    gemm(r0, Wb + wstride, accum, Nn, Din, Dout);
    float* rot[3] = {r0, r1, r2};
    const float* Tm2 = in;
    float* Tm1 = r0;
    for (int k = 2; k < Kc; ++k) {
      float* tgt = rot[(k - 1) % 3];
      neg_copy_kernel<<<gcount(nT, 256), 256, 0, stream>>>(Tm2, tgt, nT);
      spmv(Tm1, tgt, 2.0f, Din);                            // Tk = 2 L^ Tk-1 - Tk-2
      gemm(tgt, Wb + (long long)k * wstride, accum, Nn, Din, Dout);
      Tm2 = Tm1;
      Tm1 = tgt;
    }
  };

  // --- graph normalization ---
  fill(deg, 0.0f, Nn);
  deg_kernel<<<gcount(Ee, 256), 256, 0, stream>>>(dstp, deg, Ee);
  dinv_kernel<<<gcount(Nn, 256), 256, 0, stream>>>(deg, dinv, Nn);
  norm_kernel<<<gcount(Ee, 256), 256, 0, stream>>>(srcp, dstp, dinv, nw, Ee);

  // --- conv1: x(128) -> H0(256), relu, no bias/bn ---
  fill(H0, 0.0f, (long long)Nn * D1c);
  cheb(x, FIN, D1c, conv1_w, (long long)FIN * D1c, H0, B1, B2, H1);
  relu_kernel<<<gcount((long long)Nn * D1c, 256), 256, 0, stream>>>(
      H0, (long long)Nn * D1c);

  // --- conv2..4: bias, relu(pre) + bn ---
  float* hcur = H0;
  for (int i = 0; i < 3; ++i) {
    float* acc = (hcur == H0) ? H1 : H0;
    bias_init_kernel<<<gcount((long long)Nn * D1c, 256), 256, 0, stream>>>(
        acc, convs_b + (long long)i * D1c, D1c - 1, (long long)Nn * D1c);
    cheb(hcur, D1c, D1c, convs_w + (long long)i * Kc * D1c * D1c,
         (long long)D1c * D1c, acc, B1, B2, hcur);
    bn(acc, D1c, bn1_g, bn1_b, /*pre=*/1, /*post=*/0);
    hcur = acc;
  }

  // --- fc1: 256 -> 128, bn2, relu(post) ---
  float* f1 = (hcur == H0) ? H1 : H0;
  bias_init_kernel<<<gcount((long long)Nn * D2c, 256), 256, 0, stream>>>(
      f1, fc1_b, D2c - 1, (long long)Nn * D2c);
  gemm(hcur, fc1_w, f1, Nn, D1c, D2c);
  bn(f1, D2c, bn2_g, bn2_b, /*pre=*/0, /*post=*/1);

  // --- fc2: 128 -> 64, bn3 ---
  bias_init_kernel<<<gcount((long long)Nn * D3c, 256), 256, 0, stream>>>(
      B1, fc2_b, D3c - 1, (long long)Nn * D3c);
  gemm(f1, fc2_w, B1, Nn, D2c, D3c);
  bn(B1, D3c, bn3_g, bn3_b, /*pre=*/0, /*post=*/0);

  // --- fc3: 64 -> 1 ---
  fc3_kernel<<<gcount(Nn, 8), 256, 0, stream>>>(B1, fc3_w, fc3_b, (float*)d_out, Nn);
}
